// UnfoldCorrelationBlock_45414984188033
// MI455X (gfx1250) — matrix operand order repaired and checked by reading, not compiled, not run on hardware
//
#include <hip/hip_runtime.h>

typedef __attribute__((ext_vector_type(16))) _Float16 v16h;
typedef __attribute__((ext_vector_type(8)))  _Float16 v8h;
typedef __attribute__((ext_vector_type(4)))  _Float16 v4h;
typedef __attribute__((ext_vector_type(8)))  float    v8f;

#define NB    4
#define CB    128
#define HB    256
#define WB    256
#define DPAD  4
#define OUTK  9
#define HWSZ  (HB * WB)         // 65536
#define CHWSZ (CB * HWSZ)

#define ROWS   12               // staged x2 rows: y0-4 .. y0+7
#define COLS   24               // staged x2 cols: x0-4 .. x0+19
#define KPITCH 40               // 32 k-halfs + 8 pad (80B pitch: 16B-aligned,
                                // 20-dword lane stride -> conflict-free b128)

// Load one 16-half B fragment (2x ds_load_b128) from LDS.
__device__ __forceinline__ v16h ldfrag(const _Float16* p) {
    const v8h lo  = *(const v8h*)p;
    const v8h hi8 = *(const v8h*)(p + 8);
    return __builtin_shufflevector(lo, hi8,
                                   0,1,2,3,4,5,6,7,8,9,10,11,12,13,14,15);
}

// Workgroup: 256 threads = 8 waves = 4 y-rows x 2 dy-groups at one
// (n, y0..y0+3, x0..x0+15) tile. Per K-step the block stages a
// 12x24x32-channel x2 slab as f16 in LDS (converted once, OOB zeroed), then
// each wave batch-loads all 10 B fragments (ds_load_b128) and runs 10 WMMAs.
__global__ __launch_bounds__(256) void corr_wmma_kernel(
    const float* __restrict__ x1, const float* __restrict__ x2,
    float* __restrict__ out)
{
    __shared__ __align__(16) _Float16 stage[ROWS][COLS][KPITCH]; // 23 KB
    __shared__ float band[8][16][25];                            // 12.8 KB

    const int tid   = threadIdx.x;
    const int lane  = tid & 31;
    const int wid   = tid >> 5;
    const int col16 = lane & 15;
    const int hi    = lane >> 4;

    const int yoff = wid & 3;            // wave's y within the block
    const int dyg  = wid >> 2;           // 0: dy 0..4, 1: dy 5..8 (+ phantom 9)

    const int x0 = (blockIdx.x & 15) << 4;
    const int y0 = ((blockIdx.x >> 4) & 63) << 2;
    const int n  = blockIdx.x >> 10;
    const int y  = y0 + yoff;

    // Block-uniform: no row/col of the staged slab is out of bounds.
    const bool interior = (y0 >= DPAD) && (y0 + 7 < HB) &&
                          (x0 >= DPAD) && (x0 + 19 < WB);

    // ---- Preload A: x1 tile (16 px x 128 ch) as 4 k-steps of v16h.
    // A k order used here (contiguous, same as B): lanes 0-15 (M=lane): elems 0-15 -> K0-15;
    // lanes 16-31: elems 0-15 -> K16-31.
    v16h a[4];
    {
        const float* pa = x1 + (size_t)n * CHWSZ + (size_t)y * WB + x0 + col16
                             + (size_t)hi * 16 * HWSZ;
        #pragma unroll
        for (int ks = 0; ks < 4; ++ks) {
            const float* p = pa + (size_t)ks * 32 * HWSZ;
            float f[16];
            #pragma unroll
            for (int e = 0; e < 16; ++e)
                f[e] = __builtin_nontemporal_load(p + (size_t)e * HWSZ);
            #pragma unroll
            for (int e = 0; e < 16; ++e)
                a[ks][e] = (_Float16)f[e];
        }
    }

    const v8f zero = {};
    v8f acc0[5], acc1[5];
    #pragma unroll
    for (int i = 0; i < 5; ++i) { acc0[i] = zero; acc1[i] = zero; }

    const float* x2n = x2 + (size_t)n * CHWSZ;

    // Rows this wave consumes: dyi -> row = yoff + dyg*5 + dyi, clamped so the
    // phantom dy=9 (dyg=1, dyi=4) reads valid LDS (result discarded at store).
    int rowv[5];
    #pragma unroll
    for (int i = 0; i < 5; ++i) {
        int r = yoff + dyg * 5 + i;
        rowv[i] = r > ROWS - 1 ? ROWS - 1 : r;
    }

    for (int ks = 0; ks < 4; ++ks) {
        __syncthreads();   // previous k-step's readers done before restaging

        // ---- Stage: 12 rows x 24 cols x 32 channels, f32 -> f16, OOB = 0.
        // 2304 4-channel tasks / 256 threads = 9 each. Loads are always
        // unconditional (clamped addresses); OOB lanes zero the *value*.
        #pragma unroll
        for (int it = 0; it < 9; ++it) {
            const int task = tid + it * 256;       // 0..2303
            const int col  = task % COLS;
            const int rem  = task / COLS;          // 0..95
            const int cg   = rem & 7;              // 4-channel group
            const int row  = rem >> 3;             // 0..11

            const int y2 = y0 - DPAD + row;
            const int xg = x0 - DPAD + col;

            v4h h;
            if (interior) {                        // block-uniform fast path
                const float* q = x2n + (size_t)(ks * 32 + cg * 4) * HWSZ
                                     + (size_t)y2 * WB + xg;
                h[0] = (_Float16)q[0 * (size_t)HWSZ];
                h[1] = (_Float16)q[1 * (size_t)HWSZ];
                h[2] = (_Float16)q[2 * (size_t)HWSZ];
                h[3] = (_Float16)q[3 * (size_t)HWSZ];
            } else {                               // border: clamp addr, mask val
                const bool ok  = (y2 >= 0) & (y2 < HB) & (xg >= 0) & (xg < WB);
                const int  y2c = y2 < 0 ? 0 : (y2 > HB - 1 ? HB - 1 : y2);
                const int  xgc = xg < 0 ? 0 : (xg > WB - 1 ? WB - 1 : xg);
                const float* q = x2n + (size_t)(ks * 32 + cg * 4) * HWSZ
                                     + (size_t)y2c * WB + xgc;
                // Unconditional loads; select on the value (v_cndmask).
                const float v0 = q[0 * (size_t)HWSZ];
                const float v1 = q[1 * (size_t)HWSZ];
                const float v2 = q[2 * (size_t)HWSZ];
                const float v3 = q[3 * (size_t)HWSZ];
                h[0] = (_Float16)(ok ? v0 : 0.f);
                h[1] = (_Float16)(ok ? v1 : 0.f);
                h[2] = (_Float16)(ok ? v2 : 0.f);
                h[3] = (_Float16)(ok ? v3 : 0.f);
            }
            *(v4h*)&stage[row][col][cg * 4] = h;   // ds_store_b64
        }
        __syncthreads();

        // ---- Compute: 5 dy per wave, branch-free, EXEC all-1s at every WMMA.
        // B k order used here (contiguous, same as A): lanes 0-15 (N=lane): K = elem 0..15; lanes
        // 16-31: K = 16+elem -> 16 contiguous halfs at [row][col][hi*16].
        // Batch-load all 10 fragments into distinct SSA values first so the
        // scheduler can keep many ds_loads in flight (partial dscnt waits).
        v16h bf0[5], bf1[5];
        #pragma unroll
        for (int dyi = 0; dyi < 5; ++dyi) {
            bf0[dyi] = ldfrag(&stage[rowv[dyi]][col16    ][hi * 16]);
            bf1[dyi] = ldfrag(&stage[rowv[dyi]][col16 + 8][hi * 16]);
        }
        #pragma unroll
        for (int dyi = 0; dyi < 5; ++dyi) {
            acc0[dyi] = __builtin_amdgcn_wmma_f32_16x16x32_f16(
                false, a[ks], false, bf0[dyi], (short)0, acc0[dyi], false, false);
            acc1[dyi] = __builtin_amdgcn_wmma_f32_16x16x32_f16(
                false, a[ks], false, bf1[dyi], (short)0, acc1[dyi], false, false);
        }
    }

    // ---- Band extraction + store. D layout: VGPR r, lane L -> M = r+8*(L>=16),
    // N = L%16. D0 -> band cols 0..15, D1 -> 8..23 (overlap identical).
    float* outp = out + (size_t)n * (OUTK * OUTK * (size_t)HWSZ)
                      + (size_t)y * WB + x0;
    #pragma unroll
    for (int dyi = 0; dyi < 5; ++dyi) {
        const int dy = dyg * 5 + dyi;              // 9 = phantom, never stored
        #pragma unroll
        for (int r = 0; r < 8; ++r)
            band[wid][r + 8 * hi][col16] = acc0[dyi][r];
        #pragma unroll
        for (int r = 0; r < 8; ++r)
            band[wid][r + 8 * hi][col16 + 8] = acc1[dyi][r];
        // Same-wave cross-lane handoff: DS ops are in-order per wave.
        __asm__ volatile("s_wait_dscnt 0" ::: "memory");

        #pragma unroll
        for (int dxp = 0; dxp < 5; ++dxp) {
            const int   dx = dxp * 2 + hi;         // lanes 0-15 even, 16-31 odd
            const float v  = band[wid][col16][col16 + dx] * (1.0f / CB);
            if ((dx < OUTK) & (dy < OUTK))         // predicated store only
                __builtin_nontemporal_store(
                    v, outp + (size_t)(dy * OUTK + dx) * HWSZ + col16);
        }
        __asm__ volatile("" ::: "memory");         // next spill after reads
    }
}

extern "C" void kernel_launch(void* const* d_in, const int* in_sizes, int n_in,
                              void* d_out, int out_size, void* d_ws, size_t ws_size,
                              hipStream_t stream) {
    const float* x1  = (const float*)d_in[0];
    const float* x2  = (const float*)d_in[1];
    float*       out = (float*)d_out;

    // 4096 blocks: 16 x-tiles * 64 y-quads * 4 batch.
    dim3 grid(4096), block(256);
    hipLaunchKernelGGL(corr_wmma_kernel, grid, block, 0, stream, x1, x2, out);
}
